// ParseridgeModel_5291399709466
// MI455X (gfx1250) — matrix operand order repaired and checked by reading, not compiled, not run on hardware
//
#include <hip/hip_runtime.h>
#include <hip/hip_bf16.h>
#include <math.h>

// ---------------------------------------------------------------------------
// Parser MLP head on MI455X (gfx1250, wave32).
//   x[4096,1000] = masked gather of 4 rows of lstm_out  -> bf16
//   trans = tanh(tanh(x W1t + b1t) W2t + b2t)   [4096,4]
//   rel   = tanh(tanh(x W1r + b1r) W2r + b2r)   [4096,50]
// Gather-bandwidth bound (~16MB useful HBM traffic); GEMM via
// v_wmma_f32_16x16x32_bf16, f32 accumulation, K padded 1000->1024, H 125->128.
// Weights are pre-transposed/padded to bf16 in d_ws (fits easily in L2).
// This revision routes weight streaming through GLOBAL (not FLAT) loads by
// keeping weight pointers traceable to kernel args (template helper instead of
// pointer arrays), decoupling LOADcnt from the DS pipe used for A fragments.
// ---------------------------------------------------------------------------

typedef __attribute__((ext_vector_type(16))) __bf16 v16bf;
typedef __attribute__((ext_vector_type(8)))  __bf16 v8bf;
typedef __attribute__((ext_vector_type(8)))  float  v8f;

#define B_BATCH 4096
#define T_TOK   64
#define D_FEAT  250
#define KPAD    1024   // 4*250 -> 1024
#define HPAD    128    // 125 -> 128
#define NT1     8      // layer-1 N tiles (128/16)
#define NTRANS  4
#define NLAB    50

__device__ __forceinline__ __bf16 to_bf16(float f) { return (__bf16)f; }

union V16 { v16bf v; struct { v8bf lo, hi; } p; };

// -------------------- weight prep: transpose + pad + f32->bf16 --------------
// ws layout (bf16 elements):
//   [0,             131072)  W1t^T  [n=0..127][k=0..1023]
//   [131072,        262144)  W1r^T
//   [262144,        264192)  W2t^T  [j=0..15 ][n=0..127]
//   [264192,        272384)  W2r^T  [j=0..63 ][n=0..127]
__global__ void prep_weights(const float* __restrict__ W1_t,
                             const float* __restrict__ W1_r,
                             const float* __restrict__ W2_t,
                             const float* __restrict__ W2_r,
                             __bf16* __restrict__ ws) {
  const int n1  = HPAD * KPAD;   // 131072
  const int n2t = 16 * HPAD;     // 2048
  const int n2r = 64 * HPAD;     // 8192
  int tid = blockIdx.x * blockDim.x + threadIdx.x;
  if (tid >= 2 * n1 + n2t + n2r) return;
  float v = 0.f;
  if (tid < 2 * n1) {
    const float* W = (tid < n1) ? W1_t : W1_r;
    int i = (tid < n1) ? tid : tid - n1;
    int n = i >> 10;          // output col 0..127
    int k = i & 1023;         // input  row 0..1023
    if (n < 125 && k < 1000) v = W[k * 125 + n];
  } else if (tid < 2 * n1 + n2t) {
    int i = tid - 2 * n1;
    int j = i >> 7, n = i & 127;
    if (j < NTRANS && n < 125) v = W2_t[n * NTRANS + j];
  } else {
    int i = tid - 2 * n1 - n2t;
    int j = i >> 7, n = i & 127;
    if (j < NLAB && n < 125) v = W2_r[n * NLAB + j];
  }
  ws[tid] = to_bf16(v);
}

// -------------------- one MLP: layer1 WMMA -> tanh -> layer2 WMMA -> tanh ---
// Inlined twice with pointers that trace directly to kernel args so the
// backend proves addrspace(1) for weights (global_load) and addrspace(3) for
// the LDS tiles (ds_load), keeping LOADcnt and DScnt independent.
template <int NOUT, int NT2, int OBASE>
__device__ __forceinline__ void run_mlp(
    const __bf16* ldsA, __bf16* ldsH,
    const __bf16* __restrict__ w1t, const float* __restrict__ b1,
    const __bf16* __restrict__ w2t, const float* __restrict__ b2,
    int row, int hi, int n16, int m0, float* __restrict__ out) {
  // ---- layer 1: [16 x 1024] x [1024 x 128], bias preloaded into C ----
  v8f acc[NT1];
  #pragma unroll
  for (int nt = 0; nt < NT1; ++nt) {
    const int n = nt * 16 + n16;
    const float bv = (n < 125) ? b1[n] : 0.f;
    #pragma unroll
    for (int i = 0; i < 8; ++i) acc[nt][i] = bv;
  }
  const __bf16* arow = ldsA + row * KPAD;
  for (int k = 0; k < 32; ++k) {            // K = 32 chunks of 32
    V16 a;                                   // A frag: ds_load_b128 x2
    const int ao = k * 32 + hi * 16;
    a.p.lo = *(const v8bf*)(arow + ao);
    a.p.hi = *(const v8bf*)(arow + ao + 8);
    #pragma unroll
    for (int nt = 0; nt < NT1; ++nt) {      // B frag: 32B contiguous K-slice
      const __bf16* brow = w1t + (size_t)(nt * 16 + n16) * KPAD + k * 32 + hi * 16;
      const v16bf bm = *(const v16bf*)brow;
      acc[nt] = __builtin_amdgcn_wmma_f32_16x16x32_bf16(
          false, a.v, false, bm, (short)0, acc[nt], false, false);
    }
  }
  // ---- tanh -> bf16 hidden tile in LDS (row-major A layout for layer 2) ----
  __syncthreads();   // prior user of ldsH done
  #pragma unroll
  for (int nt = 0; nt < NT1; ++nt) {
    #pragma unroll
    for (int i = 0; i < 8; ++i) {
      const int m = i + hi * 8;              // C-layout: vgpr i, lane-hi -> row
      ldsH[m * HPAD + nt * 16 + n16] = to_bf16(tanhf(acc[nt][i]));
    }
  }
  __syncthreads();
  // ---- layer 2: [16 x 128] x [128 x (NT2*16)], K = 4 chunks of 32 ----
  #pragma unroll
  for (int nt2 = 0; nt2 < NT2; ++nt2) {
    v8f acc2;
    const int n = nt2 * 16 + n16;
    const float bv = (n < NOUT) ? b2[n] : 0.f;
    #pragma unroll
    for (int i = 0; i < 8; ++i) acc2[i] = bv;
    const __bf16* hrow = ldsH + row * HPAD;
    #pragma unroll
    for (int k = 0; k < 4; ++k) {
      V16 a;
      const int ao = k * 32 + hi * 16;
      a.p.lo = *(const v8bf*)(hrow + ao);
      a.p.hi = *(const v8bf*)(hrow + ao + 8);
      const __bf16* brow = w2t + (size_t)(nt2 * 16 + n16) * HPAD + k * 32 + hi * 16;
      const v16bf bm = *(const v16bf*)brow;
      acc2 = __builtin_amdgcn_wmma_f32_16x16x32_bf16(
          false, a.v, false, bm, (short)0, acc2, false, false);
    }
    if (n < NOUT) {                          // guard only stores; EXEC full at WMMA
      #pragma unroll
      for (int i = 0; i < 8; ++i) {
        const int m = i + hi * 8;
        out[OBASE + (size_t)(m0 + m) * NOUT + n] = tanhf(acc2[i]);
      }
    }
  }
}

// -------------------- fused gather + 2x (GEMM, tanh, GEMM, tanh) ------------
__global__ __launch_bounds__(32)
void parser_mlp_kernel(const float* __restrict__ lstm_out,
                       const int*   __restrict__ stack_index,
                       const int*   __restrict__ stack_len,
                       const int*   __restrict__ buffer_index,
                       const int*   __restrict__ buffer_len,
                       const float* __restrict__ b1_t, const float* __restrict__ b2_t,
                       const float* __restrict__ b1_r, const float* __restrict__ b2_r,
                       const __bf16* __restrict__ ws,
                       float* __restrict__ out) {
  __shared__ __align__(32) __bf16 ldsA[16 * KPAD];  // 32KB gathered A tile (bf16)
  __shared__ __align__(32) __bf16 ldsH[16 * HPAD];  // 4KB  hidden tile

  const int lane = threadIdx.x;   // 0..31 (wave32)
  const int row  = lane & 15;     // M within tile / A-row for this lane
  const int hi   = lane >> 4;     // lane-hi half selector (WMMA interleave)
  const int n16  = lane & 15;     // N within 16-wide tile for B/C/D operands
  const int m0   = blockIdx.x * 16;
  const int b    = m0 + row;

  // ---- gather + mask + f32->bf16 into LDS A[16][1024] (2 lanes per row) ----
  const int slen = stack_len[b];
  const int blen = buffer_len[b];
  const int cbase = hi * 125;     // this lane covers cols [cbase, cbase+125)
  #pragma unroll
  for (int seg = 0; seg < 4; ++seg) {
    const int idx   = (seg < 3) ? stack_index[b * 3 + seg] : buffer_index[b];
    const int valid = (seg < 3) ? (seg < slen) : (0 < blen);
    const float* src = lstm_out + ((size_t)b * T_TOK + idx) * D_FEAT;
    __bf16* dst = ldsA + row * KPAD + seg * D_FEAT;
    for (int j = 0; j < 125; ++j) {
      const int c = cbase + j;
      dst[c] = to_bf16(valid ? src[c] : 0.f);
    }
  }
  { // zero K padding cols 1000..1023
    __bf16* dst = ldsA + row * KPAD + 1000 + hi * 12;
    for (int j = 0; j < 12; ++j) dst[j] = to_bf16(0.f);
  }
  __syncthreads();

  // transitions MLP: N=4 (1 tile), then relations MLP: N=50 (4 tiles)
  run_mlp<NTRANS, 1, 0>(ldsA, ldsH,
                        ws, b1_t,
                        ws + 2 * HPAD * KPAD, b2_t,
                        row, hi, n16, m0, out);
  run_mlp<NLAB, 4, B_BATCH * NTRANS>(ldsA, ldsH,
                                     ws + HPAD * KPAD, b1_r,
                                     ws + 2 * HPAD * KPAD + 16 * HPAD, b2_r,
                                     row, hi, n16, m0, out);
}

// ---------------------------------------------------------------------------
extern "C" void kernel_launch(void* const* d_in, const int* in_sizes, int n_in,
                              void* d_out, int out_size, void* d_ws, size_t ws_size,
                              hipStream_t stream) {
  const float* lstm_out     = (const float*)d_in[0];
  const int*   stack_index  = (const int*)d_in[1];
  const int*   stack_len    = (const int*)d_in[2];
  const int*   buffer_index = (const int*)d_in[3];
  const int*   buffer_len   = (const int*)d_in[4];
  const float* W1_t = (const float*)d_in[5];
  const float* b1_t = (const float*)d_in[6];
  const float* W2_t = (const float*)d_in[7];
  const float* b2_t = (const float*)d_in[8];
  const float* W1_r = (const float*)d_in[9];
  const float* b1_r = (const float*)d_in[10];
  const float* W2_r = (const float*)d_in[11];
  const float* b2_r = (const float*)d_in[12];
  __bf16* ws = (__bf16*)d_ws;   // needs 272384 bf16 = 544768 bytes

  const int total = 2 * HPAD * KPAD + 16 * HPAD + 64 * HPAD;
  prep_weights<<<(total + 255) / 256, 256, 0, stream>>>(W1_t, W1_r, W2_t, W2_r, ws);

  parser_mlp_kernel<<<B_BATCH / 16, 32, 0, stream>>>(
      lstm_out, stack_index, stack_len, buffer_index, buffer_len,
      b1_t, b2_t, b1_r, b2_r, ws, (float*)d_out);
}
